// DeformableTransformerEncoderLayer_31714038514180
// MI455X (gfx1250) — compile-verified, matrix operand order checked by reading, not run on hardware
//
#include <hip/hip_runtime.h>
#include <hip/hip_bf16.h>
#include <cstdint>

// ---------------------------------------------------------------------------
// DeformableTransformerEncoderLayer for MI455X (gfx1250, wave32, WMMA + TDM)
//   N=8, Lq=5440, T=N*Lq=43520, D=256, NH=8, HD=32, NL=4, NP=4, DFF=1024
//
// Bandwidth-bound (~71 GFLOP GEMM, ~0.5 GB traffic). All six GEMMs use
// v_wmma_f32_16x16x32_f16 (f32 acc). Weights pre-transposed to (N,K) f16.
// GEMM tiles are staged global->LDS by the Tensor Data Mover (one DMA per
// block issued by wave 0, TENSORcnt-tracked, LDS rows padded 64B+16B by the
// TDM pad feature); fallback is register-staged b128 copies if the TDM
// builtin is unavailable. LDS is double-buffered, one barrier per K-step.
// ---------------------------------------------------------------------------

typedef __attribute__((ext_vector_type(16))) _Float16 v16h;
typedef __attribute__((ext_vector_type(8)))  float    v8f;
typedef __attribute__((ext_vector_type(4)))  unsigned u32x4;
typedef __attribute__((ext_vector_type(8)))  int      i32x8;
typedef __attribute__((ext_vector_type(4)))  int      i32x4;

union Frag16 { v16h v; unsigned u[8]; };

#define NTOK   43520
#define DMODEL 256
#define NHEAD  8
#define HDIM   32
#define NLVL   4
#define NPTS   4
#define LQ     5440

#if defined(__HIP_DEVICE_COMPILE__) && __has_builtin(__builtin_amdgcn_tensor_load_to_lds)
#define USE_TDM 1
#else
#define USE_TDM 0
#endif

// ------------------------------ prep ---------------------------------------
__global__ void prep_kernel(const float* __restrict__ src,
                            const float* __restrict__ pos,
                            _Float16* __restrict__ s16,
                            _Float16* __restrict__ q16,
                            size_t n) {
  size_t i = (size_t)blockIdx.x * blockDim.x + threadIdx.x;
  size_t stride = (size_t)gridDim.x * blockDim.x;
  for (; i < n; i += stride) {
    float s = src[i];
    s16[i] = (_Float16)s;
    q16[i] = (_Float16)(s + pos[i]);
  }
}

// Transpose + convert: Wt[n*K + k] = (f16) W[k*N + n].
__global__ __launch_bounds__(256)
void cvt_transpose_kernel(const float* __restrict__ W,
                          _Float16* __restrict__ Wt, int K, int N) {
  int n = blockIdx.x * 32 + (threadIdx.x & 31);
  int k = blockIdx.y * 8 + (threadIdx.x >> 5);
  if (n < N && k < K)
    Wt[(size_t)n * K + k] = (_Float16)W[(size_t)k * N + n];
}

// --------------------------- TDM descriptor helper -------------------------
#if USE_TDM
// 2D tile load Global->LDS per CDNA5 ISA D# (groups 0/1; groups 2/3 zero for
// <=2D). data_size=2B. LDS rows padded: 16 DWORDs data + 4 DWORDs pad (80B),
// matching the LDA/LDB=40-half fragment layout.
__device__ __forceinline__ void tdm_load_tile(const _Float16* gsrc,
                                              unsigned lds_byte,
                                              int K, int rows) {
  unsigned long long ga = (unsigned long long)(uintptr_t)gsrc;
  u32x4 g0;
  g0.x = 0x1u;                                        // count=1 (valid), user
  g0.y = lds_byte;                                    // lds_addr [63:32]
  g0.z = (unsigned)(ga & 0xffffffffu);                // global_addr[31:0]
  g0.w = (unsigned)((ga >> 32) & 0x01ffffffu)         // global_addr[56:32]
       | (2u << 30);                                  // type=2 ("image")
  i32x8 g1;
  g1[0] = (1 << 16)                                   // data_size: 2 bytes
        | (1 << 20)                                   // pad_enable
        | (3 << 22)                                   // pad_interval: 16 dw
        | (3 << 25);                                  // pad_amount: 4 dw
  g1[1] = (K & 0xffff) << 16;                         // tensor_dim0[15:0]
  g1[2] = ((K >> 16) & 0xffff) | ((rows & 0xffff) << 16);   // dim0 hi, dim1 lo
  g1[3] = ((rows >> 16) & 0xffff) | (32 << 16);       // dim1 hi, tile_dim0=BK
  g1[4] = rows & 0xffff;                              // tile_dim1; tile_dim2=0
  g1[5] = K;                                          // dim0_stride[31:0]
  g1[6] = 0;                                          // stride hi, dim1_stride
  g1[7] = 0;
  i32x4 z4 = {0, 0, 0, 0};
#if __clang_major__ >= 23
  i32x8 z8 = {0, 0, 0, 0, 0, 0, 0, 0};
  __builtin_amdgcn_tensor_load_to_lds(g0, g1, z4, z4, z8, 0);
#else
  __builtin_amdgcn_tensor_load_to_lds(g0, g1, z4, z4, 0);
#endif
}

__device__ __forceinline__ unsigned lds_off(const void* p) {
  return (unsigned)(unsigned long long)(uintptr_t)p;   // low 32 bits give the LDS byte offset
}
#endif

// ------------------------------ GEMM (WMMA) --------------------------------
// C[M,Nn] = A[M,K] * Bt[Nn,K]^T + bias (opt relu); f16 in, f32 acc.
// Block: 256 threads = 8 waves, tile 64x128, waves 2(M) x 4(N), 32x32 each.
#define BM 64
#define BN 128
#define BK 32
#define LDA 40   // halves; 80B rows: 16B-aligned b128, conflict-spread banks
#define LDB 40

template <int RELU, int OUT16>
__global__ __launch_bounds__(256)
void gemm_wmma_kernel(const _Float16* __restrict__ A,
                      const _Float16* __restrict__ Bt,
                      const float* __restrict__ bias,
                      float* __restrict__ Cf32,
                      _Float16* __restrict__ Cf16,
                      int K, int Nn) {
  __shared__ _Float16 sA[2][BM * LDA];   // [buf][m][k]
  __shared__ _Float16 sB[2][BN * LDB];   // [buf][n][k]

  const int tid  = threadIdx.x;
  const int lane = tid & 31;
  const int wid  = tid >> 5;
  const int wm   = wid & 1;
  const int wn   = wid >> 1;
  const int m0   = blockIdx.x * BM;
  const int n0   = blockIdx.y * BN;

  const int halfsel = (lane >> 4) & 1;
  const int l15     = lane & 15;

  const _Float16* aBase = A  + (size_t)m0 * K;
  const _Float16* bBase = Bt + (size_t)n0 * K;

  v8f acc[2][2] = {};

#if USE_TDM
  const bool issuer = (wid == 0);
  if (issuer) {
    tdm_load_tile(aBase, lds_off(&sA[0][0]), K, BM);
    tdm_load_tile(bBase, lds_off(&sB[0][0]), K, BN);
    __builtin_amdgcn_s_wait_tensorcnt(0);
  }
  __syncthreads();
#else
  const int srow = tid >> 2;             // 0..63
  const int scol = (tid & 3) << 3;       // halves: 0,8,16,24
  const _Float16* aPtr  = aBase + (size_t)srow * K + scol;
  const _Float16* bPtr0 = bBase + (size_t)srow * K + scol;
  const _Float16* bPtr1 = bBase + (size_t)(64 + srow) * K + scol;
  uint4 ra  = *(const uint4*)(aPtr);
  uint4 rb0 = *(const uint4*)(bPtr0);
  uint4 rb1 = *(const uint4*)(bPtr1);
  *(uint4*)(&sA[0][srow * LDA + scol]) = ra;
  *(uint4*)(&sB[0][srow * LDB + scol]) = rb0;
  *(uint4*)(&sB[0][(64 + srow) * LDB + scol]) = rb1;
  __syncthreads();
#endif

  int cur = 0;
  for (int k0 = 0; k0 < K; k0 += BK) {
    const int k1 = k0 + BK;
    const bool more = (k1 < K);
    const int nxt = cur ^ 1;

#if USE_TDM
    if (more && issuer) {                 // DMA next tile while computing
      tdm_load_tile(aBase + k1, lds_off(&sA[nxt][0]), K, BM);
      tdm_load_tile(bBase + k1, lds_off(&sB[nxt][0]), K, BN);
    }
#else
    if (more) {
      ra  = *(const uint4*)(aPtr  + k1);
      rb0 = *(const uint4*)(bPtr0 + k1);
      rb1 = *(const uint4*)(bPtr1 + k1);
      __builtin_prefetch(aPtr + k1 + BK, 0, 1);
    }
#endif

    // fragments per 16-bit A-matrix layout: VGPR r holds K pair
    // 16*(r>=4) + 8*(lane>=16) + 2*(r&3); B tile is [n][k] so symmetric.
    Frag16 fa[2], fb[2];
    #pragma unroll
    for (int t = 0; t < 2; ++t) {
      const int row = wm * 32 + t * 16 + l15;
      const int col = wn * 32 + t * 16 + l15;
      #pragma unroll
      for (int r = 0; r < 8; ++r) {
        const int kk = ((r >= 4) ? 16 : 0) + halfsel * 8 + ((r & 3) << 1);
        fa[t].u[r] = *(const unsigned*)(&sA[cur][row * LDA + kk]);
        fb[t].u[r] = *(const unsigned*)(&sB[cur][col * LDB + kk]);
      }
    }
    #pragma unroll
    for (int mt = 0; mt < 2; ++mt)
      #pragma unroll
      for (int nt = 0; nt < 2; ++nt)
        acc[mt][nt] = __builtin_amdgcn_wmma_f32_16x16x32_f16(
            false, fa[mt].v, false, fb[nt].v, (short)0, acc[mt][nt],
            false, false);

#if USE_TDM
    if (more && issuer) __builtin_amdgcn_s_wait_tensorcnt(0);
#else
    if (more) {
      *(uint4*)(&sA[nxt][srow * LDA + scol]) = ra;
      *(uint4*)(&sB[nxt][srow * LDB + scol]) = rb0;
      *(uint4*)(&sB[nxt][(64 + srow) * LDB + scol]) = rb1;
    }
#endif
    __syncthreads();
    cur = nxt;
  }

  // epilogue: C/D layout -> VGPR r holds M=r (lanes 0-15) / M=8+r (16-31)
  const int mhi = (lane >> 4) << 3;
  #pragma unroll
  for (int mt = 0; mt < 2; ++mt) {
    #pragma unroll
    for (int nt = 0; nt < 2; ++nt) {
      const int gn = n0 + wn * 32 + nt * 16 + l15;
      const float bv = bias[gn];
      #pragma unroll
      for (int r = 0; r < 8; ++r) {
        const int gm = m0 + wm * 32 + mt * 16 + mhi + r;
        float v = acc[mt][nt][r] + bv;
        if (RELU) v = fmaxf(v, 0.0f);
        const size_t o = (size_t)gm * Nn + gn;
        if (OUT16) Cf16[o] = (_Float16)v;
        else       Cf32[o] = v;
      }
    }
  }
}

// ------------------------- deformable sampler ------------------------------
// One wave32 per (n, q, h); lane = channel d of HD=32. Softmax over NL*NP=16
// computed redundantly per lane; 16 bilinear taps gather 128B rows (coalesced).
__global__ __launch_bounds__(256)
void msda_sample_kernel(const float* __restrict__ value,   // (T,256)
                        const float* __restrict__ off,     // (T,256)
                        const float* __restrict__ attnL,   // (T,128) logits
                        const float* __restrict__ refp,    // (N,Lq,NL,2)
                        _Float16* __restrict__ out16) {    // (T,256)
  const int gw   = blockIdx.x * 8 + (threadIdx.x >> 5);
  const int lane = threadIdx.x & 31;
  if (gw >= NTOK * NHEAD) return;
  const int h = gw & (NHEAD - 1);
  const int t = gw >> 3;
  const int n = t / LQ;

  constexpr int lvlH[NLVL] = {64, 32, 16, 8};
  constexpr int lvlW[NLVL] = {64, 32, 16, 8};
  constexpr int lvlS[NLVL] = {0, 4096, 5120, 5376};

  const float* arow = attnL + (size_t)t * 128 + h * 16;
  float w[16];
  float mx = -3.0e38f;
  #pragma unroll
  for (int i = 0; i < 16; ++i) { w[i] = arow[i]; mx = fmaxf(mx, w[i]); }
  float ssum = 0.0f;
  #pragma unroll
  for (int i = 0; i < 16; ++i) { w[i] = __expf(w[i] - mx); ssum += w[i]; }
  const float inv = 1.0f / ssum;

  const float* orow  = off + (size_t)t * 256 + h * 32;
  const float* vbase = value + (size_t)n * LQ * 256 + h * HDIM + lane;

  float acc = 0.0f;
  #pragma unroll
  for (int l = 0; l < NLVL; ++l) {
    const int H = lvlH[l], W = lvlW[l], S = lvlS[l];
    const float rx = refp[((size_t)t * NLVL + l) * 2 + 0];
    const float ry = refp[((size_t)t * NLVL + l) * 2 + 1];
    #pragma unroll
    for (int p = 0; p < NPTS; ++p) {
      const int pi = l * NPTS + p;
      const float aw = w[pi] * inv;
      const float lx = rx + orow[pi * 2 + 0] / (float)W;
      const float ly = ry + orow[pi * 2 + 1] / (float)H;
      const float x = lx * (float)W - 0.5f;
      const float y = ly * (float)H - 0.5f;
      const float x0f = floorf(x), y0f = floorf(y);
      const int x0 = (int)x0f, y0 = (int)y0f;
      const float wx1 = x - x0f, wy1 = y - y0f;
      const float wx0 = 1.0f - wx1, wy0 = 1.0f - wy1;
      float v00 = 0.0f, v01 = 0.0f, v10 = 0.0f, v11 = 0.0f;
      if (y0 >= 0 && y0 < H) {
        if (x0 >= 0 && x0 < W)         v00 = vbase[((size_t)(S + y0 * W + x0)) * 256];
        if (x0 + 1 >= 0 && x0 + 1 < W) v01 = vbase[((size_t)(S + y0 * W + x0 + 1)) * 256];
      }
      if (y0 + 1 >= 0 && y0 + 1 < H) {
        if (x0 >= 0 && x0 < W)         v10 = vbase[((size_t)(S + (y0 + 1) * W + x0)) * 256];
        if (x0 + 1 >= 0 && x0 + 1 < W) v11 = vbase[((size_t)(S + (y0 + 1) * W + x0 + 1)) * 256];
      }
      acc += aw * (wy0 * (wx0 * v00 + wx1 * v01) + wy1 * (wx0 * v10 + wx1 * v11));
    }
  }
  out16[(size_t)t * 256 + h * HDIM + lane] = (_Float16)acc;
}

// -------------------------- residual + LayerNorm ---------------------------
__global__ __launch_bounds__(256)
void ln_fuse_kernel(const float* __restrict__ a,
                    const float* __restrict__ b,
                    const float* __restrict__ g,
                    const float* __restrict__ be,
                    float* __restrict__ outf,
                    _Float16* __restrict__ outh) {
  const int t = blockIdx.x;
  const int i = threadIdx.x;
  const size_t o = (size_t)t * DMODEL + i;
  float v = a[o] + b[o];

  __shared__ float red[8];
  __shared__ float stat[2];

  float s = v;
  #pragma unroll
  for (int d = 16; d > 0; d >>= 1) s += __shfl_down(s, d, 32);
  if ((i & 31) == 0) red[i >> 5] = s;
  __syncthreads();
  if (i == 0) {
    float tot = 0.0f;
    #pragma unroll
    for (int k = 0; k < 8; ++k) tot += red[k];
    stat[0] = tot * (1.0f / (float)DMODEL);
  }
  __syncthreads();
  const float mean = stat[0];
  const float d0 = v - mean;
  float s2 = d0 * d0;
  #pragma unroll
  for (int d = 16; d > 0; d >>= 1) s2 += __shfl_down(s2, d, 32);
  if ((i & 31) == 0) red[i >> 5] = s2;
  __syncthreads();
  if (i == 0) {
    float tot = 0.0f;
    #pragma unroll
    for (int k = 0; k < 8; ++k) tot += red[k];
    stat[1] = rsqrtf(tot * (1.0f / (float)DMODEL) + 1e-5f);
  }
  __syncthreads();
  const float y = d0 * stat[1] * g[i] + be[i];
  outf[o] = y;
  if (outh) outh[o] = (_Float16)y;
}

// ------------------------------ launch -------------------------------------
extern "C" void kernel_launch(void* const* d_in, const int* in_sizes, int n_in,
                              void* d_out, int out_size, void* d_ws, size_t ws_size,
                              hipStream_t stream) {
  (void)in_sizes; (void)n_in; (void)out_size; (void)ws_size;

  const float* src   = (const float*)d_in[0];
  const float* pos   = (const float*)d_in[1];
  const float* refp  = (const float*)d_in[2];
  // d_in[3] = spatial_shapes (static, hardcoded)
  const float* W_val = (const float*)d_in[4];
  const float* b_val = (const float*)d_in[5];
  const float* W_off = (const float*)d_in[6];
  const float* b_off = (const float*)d_in[7];
  const float* W_att = (const float*)d_in[8];
  const float* b_att = (const float*)d_in[9];
  const float* W_out = (const float*)d_in[10];
  const float* b_out = (const float*)d_in[11];
  const float* ln1_g = (const float*)d_in[12];
  const float* ln1_b = (const float*)d_in[13];
  const float* W1    = (const float*)d_in[14];
  const float* b1    = (const float*)d_in[15];
  const float* W2    = (const float*)d_in[16];
  const float* b2    = (const float*)d_in[17];
  const float* ln2_g = (const float*)d_in[18];
  const float* ln2_b = (const float*)d_in[19];

  const size_t T = NTOK;
  char* ws = (char*)d_ws;
  size_t cur = 0;
  auto alloc = [&](size_t bytes) -> void* {
    size_t a = (cur + 255) & ~(size_t)255;
    cur = a + bytes;
    return (void*)(ws + a);
  };

  _Float16* s16    = (_Float16*)alloc(T * 256 * 2);
  _Float16* q16    = (_Float16*)alloc(T * 256 * 2);
  _Float16* wv16   = (_Float16*)alloc(65536 * 2);
  _Float16* woff16 = (_Float16*)alloc(65536 * 2);
  _Float16* watt16 = (_Float16*)alloc(32768 * 2);
  _Float16* wout16 = (_Float16*)alloc(65536 * 2);
  _Float16* w1_16  = (_Float16*)alloc(262144 * 2);
  _Float16* w2_16  = (_Float16*)alloc(262144 * 2);
  float*    valf   = (float*)alloc(T * 256 * 4);
  float*    offf   = (float*)alloc(T * 256 * 4);
  float*    attf   = (float*)alloc(T * 128 * 4);
  _Float16* msda16 = (_Float16*)alloc(T * 256 * 2);
  float*    src2f  = (float*)alloc(T * 256 * 4);
  float*    xf     = (float*)alloc(T * 256 * 4);
  _Float16* x16    = (_Float16*)alloc(T * 256 * 2);
  _Float16* h16    = (_Float16*)alloc(T * 1024 * 2);
  float*    ffn2f  = (float*)alloc(T * 256 * 4);

  prep_kernel<<<2048, 256, 0, stream>>>(src, pos, s16, q16, T * 256);

  cvt_transpose_kernel<<<dim3(8, 32),  256, 0, stream>>>(W_val, wv16, 256, 256);
  cvt_transpose_kernel<<<dim3(8, 32),  256, 0, stream>>>(W_off, woff16, 256, 256);
  cvt_transpose_kernel<<<dim3(4, 32),  256, 0, stream>>>(W_att, watt16, 256, 128);
  cvt_transpose_kernel<<<dim3(8, 32),  256, 0, stream>>>(W_out, wout16, 256, 256);
  cvt_transpose_kernel<<<dim3(32, 32), 256, 0, stream>>>(W1, w1_16, 256, 1024);
  cvt_transpose_kernel<<<dim3(8, 128), 256, 0, stream>>>(W2, w2_16, 1024, 256);

  const int mb = NTOK / BM;  // 680
  gemm_wmma_kernel<0, 0><<<dim3(mb, 256 / BN), 256, 0, stream>>>(
      s16, wv16, b_val, valf, nullptr, 256, 256);
  gemm_wmma_kernel<0, 0><<<dim3(mb, 256 / BN), 256, 0, stream>>>(
      q16, woff16, b_off, offf, nullptr, 256, 256);
  gemm_wmma_kernel<0, 0><<<dim3(mb, 128 / BN), 256, 0, stream>>>(
      q16, watt16, b_att, attf, nullptr, 256, 128);

  msda_sample_kernel<<<NTOK, 256, 0, stream>>>(valf, offf, attf, refp, msda16);

  gemm_wmma_kernel<0, 0><<<dim3(mb, 256 / BN), 256, 0, stream>>>(
      msda16, wout16, b_out, src2f, nullptr, 256, 256);

  ln_fuse_kernel<<<NTOK, 256, 0, stream>>>(src, src2f, ln1_g, ln1_b, xf, x16);

  gemm_wmma_kernel<1, 1><<<dim3(mb, 1024 / BN), 256, 0, stream>>>(
      x16, w1_16, b1, nullptr, h16, 256, 1024);
  gemm_wmma_kernel<0, 0><<<dim3(mb, 256 / BN), 256, 0, stream>>>(
      h16, w2_16, b2, ffn2f, nullptr, 1024, 256);

  ln_fuse_kernel<<<NTOK, 256, 0, stream>>>(xf, ffn2f, ln2_g, ln2_b,
                                           (float*)d_out, nullptr);
}
